// PointnetSAModuleMSG_3169685865252
// MI455X (gfx1250) — matrix operand order repaired and checked by reading, not compiled, not run on hardware
//
#include <hip/hip_runtime.h>
#include <hip/hip_bf16.h>

typedef __attribute__((ext_vector_type(16))) _Float16 v16h;
typedef __attribute__((ext_vector_type(8)))  _Float16 v8h;
typedef __attribute__((ext_vector_type(8)))  float    v8f;

#define B_      4
#define N_      16384
#define C_      64
#define NPOINT_ 2048
#define BN_EPS_ 1e-5f

// ---------------------------------------------------------------------------
// Kernel 1: Furthest Point Sampling. One workgroup per batch.
// CDNA5: 320 KB LDS/WGP -> keep BOTH xyz (192 KB, SoA) and dist (64 KB)
// fully LDS-resident; the 2047-round serial loop never touches global memory
// except to emit the selected point.
// ---------------------------------------------------------------------------
__global__ __launch_bounds__(1024) void fps_kernel(
    const float* __restrict__ xyz, float* __restrict__ new_xyz,
    int* __restrict__ fps_idx)
{
    __shared__ float sPX[N_];            // 64 KB
    __shared__ float sPY[N_];            // 64 KB
    __shared__ float sPZ[N_];            // 64 KB
    __shared__ float sDist[N_];          // 64 KB running min-dist^2
    __shared__ float sRedV[32];
    __shared__ int   sRedI[32];
    __shared__ float sCur[3];

    const int b   = blockIdx.x;
    const int tid = threadIdx.x;
    const int nt  = blockDim.x;
    const float* px = xyz + (size_t)b * N_ * 3;

    const float x0 = px[0], y0 = px[1], z0 = px[2];
    for (int i = tid; i < N_; i += nt) {
        float x = px[i*3+0], y = px[i*3+1], z = px[i*3+2];
        sPX[i] = x; sPY[i] = y; sPZ[i] = z;
        float dx = x-x0, dy = y-y0, dz = z-z0;
        sDist[i] = dx*dx + dy*dy + dz*dz;
    }
    if (tid == 0) {
        fps_idx[(size_t)b*NPOINT_ + 0] = 0;
        new_xyz[((size_t)b*NPOINT_ + 0)*3 + 0] = x0;
        new_xyz[((size_t)b*NPOINT_ + 0)*3 + 1] = y0;
        new_xyz[((size_t)b*NPOINT_ + 0)*3 + 2] = z0;
        sCur[0] = x0; sCur[1] = y0; sCur[2] = z0;
    }
    __syncthreads();

    const int lane = tid & 31, wv = tid >> 5;
    for (int it = 1; it < NPOINT_; ++it) {
        const float cx = sCur[0], cy = sCur[1], cz = sCur[2];
        float bestV = -1.0f; int bestI = 0;
        for (int i = tid; i < N_; i += nt) {
            float dx = sPX[i]-cx, dy = sPY[i]-cy, dz = sPZ[i]-cz;
            float d  = dx*dx + dy*dy + dz*dz;
            float dd = fminf(sDist[i], d);
            sDist[i] = dd;
            if (dd > bestV) { bestV = dd; bestI = i; }
        }
        #pragma unroll
        for (int off = 16; off > 0; off >>= 1) {
            float ov = __shfl_down(bestV, off, 32);
            int   oi = __shfl_down(bestI, off, 32);
            if (ov > bestV) { bestV = ov; bestI = oi; }
        }
        if (lane == 0) { sRedV[wv] = bestV; sRedI[wv] = bestI; }
        __syncthreads();
        if (wv == 0) {
            bestV = sRedV[lane]; bestI = sRedI[lane];
            #pragma unroll
            for (int off = 16; off > 0; off >>= 1) {
                float ov = __shfl_down(bestV, off, 32);
                int   oi = __shfl_down(bestI, off, 32);
                if (ov > bestV) { bestV = ov; bestI = oi; }
            }
            if (lane == 0) {
                fps_idx[(size_t)b*NPOINT_ + it] = bestI;
                float nx = sPX[bestI], ny = sPY[bestI], nz = sPZ[bestI];
                new_xyz[((size_t)b*NPOINT_ + it)*3 + 0] = nx;
                new_xyz[((size_t)b*NPOINT_ + it)*3 + 1] = ny;
                new_xyz[((size_t)b*NPOINT_ + it)*3 + 2] = nz;
                sCur[0] = nx; sCur[1] = ny; sCur[2] = nz;
            }
        }
        __syncthreads();
    }
}

// ---------------------------------------------------------------------------
// Kernel 2: Ball query — one wave32 per query point. Each step tests 32
// candidates; ballot + prefix-popcount appends hits in ascending index
// order; early exit once NS found. xyz is L2-resident (192 KB/batch).
// ---------------------------------------------------------------------------
template<int NS>
__global__ __launch_bounds__(256) void ball_query_kernel(
    const float* __restrict__ xyz, const float* __restrict__ new_xyz,
    int* __restrict__ idx, float radius)
{
    const int gw   = (blockIdx.x * blockDim.x + threadIdx.x) >> 5;  // wave id
    const int lane = threadIdx.x & 31;
    if (gw >= B_ * NPOINT_) return;
    const int b = gw / NPOINT_;
    const float* px = xyz + (size_t)b * N_ * 3;
    const float cx = new_xyz[(size_t)gw*3+0];
    const float cy = new_xyz[(size_t)gw*3+1];
    const float cz = new_xyz[(size_t)gw*3+2];
    const float r2 = radius * radius;
    int* o = idx + (size_t)gw * NS;

    int cnt = 0, first = -1;
    for (int base = 0; base < N_ && cnt < NS; base += 32) {
        const int i = base + lane;
        float dx = px[i*3+0]-cx, dy = px[i*3+1]-cy, dz = px[i*3+2]-cz;
        const bool hit = (dx*dx + dy*dy + dz*dz) < r2;
        const unsigned mask = (unsigned)__ballot(hit);
        if (hit) {
            int pos = cnt + __popc(mask & ((1u << lane) - 1u));
            if (pos < NS) o[pos] = i;
        }
        if (first < 0 && mask) first = base + __ffs(mask) - 1;
        cnt += __popc(mask);
    }
    if (cnt > NS) cnt = NS;
    if (first < 0) first = 0;
    for (int p = cnt + lane; p < NS; p += 32) o[p] = first;
}

// ---------------------------------------------------------------------------
// Kernel 2.5: one-time weight prep. f32 W (Co x Ci) -> f16, K padded to
// stride 128 with zeros; BN folded: scale = g*rsqrt(1+eps), bias = b.
// ---------------------------------------------------------------------------
__global__ void prep_weights_kernel(
    const float* __restrict__ W, const float* __restrict__ g,
    const float* __restrict__ b, int Co, int Ci,
    _Float16* __restrict__ dW, float* __restrict__ dScale,
    float* __restrict__ dBias)
{
    const int e = blockIdx.x * blockDim.x + threadIdx.x;
    if (e < Co * 128) {
        const int m = e >> 7, k = e & 127;
        dW[e] = (_Float16)((k < Ci) ? W[m*Ci + k] : 0.0f);
    }
    if (e < Co) {
        dScale[e] = g[e] * rsqrtf(1.0f + BN_EPS_);
        dBias[e]  = b[e];
    }
}

// ---------------------------------------------------------------------------
// WMMA MLP layer: Y[m,col] = relu(scale[m]*(W·X)[m,col] + bias[m]), f16 out.
// Activations column-major [col][k] (stride 128 halves) -> contiguous
// 16-half B-fragments. Weights row-major [m][k] (stride 128 halves).
// ---------------------------------------------------------------------------
__device__ __forceinline__ void mlp_layer(
    const _Float16* __restrict__ sXin, _Float16* __restrict__ sYout,
    const _Float16* __restrict__ sW,
    const float* __restrict__ sScale, const float* __restrict__ sBias,
    int Co, int K, int lane, int wv)
{
    const int ntiles = (Co >> 4) * 4;            // (Co/16) x (64/16) tiles
    for (int t = wv; t < ntiles; t += 8) {
        const int mt = t >> 2, ct = t & 3;
        const int m     = (mt << 4) + (lane & 15);
        const int koffA = (lane >> 4) << 4;      // 0 or 16
        const int n     = (ct << 4) + (lane & 15);
        const int koffB = (lane >> 4) << 4;      // 0 or 16
        v8f c = {};
        for (int kk = 0; kk < K; kk += 32) {
            v8h a0 = *(const v8h*)&sW[m * 128 + kk + koffA];
            v8h a1 = *(const v8h*)&sW[m * 128 + kk + 8 + koffA];
            v16h a = __builtin_shufflevector(a0, a1,
                        0,1,2,3,4,5,6,7,8,9,10,11,12,13,14,15);
            v16h bf = *(const v16h*)&sXin[n * 128 + kk + koffB];
            c = __builtin_amdgcn_wmma_f32_16x16x32_f16(
                    false, a, false, bf, (short)0, c, false, false);
        }
        const int mrow = (mt << 4) + ((lane >> 4) << 3);
        const int col  = (ct << 4) + (lane & 15);
        #pragma unroll
        for (int v = 0; v < 8; ++v) {
            const int mm = mrow + v;
            float val = fmaxf(0.0f, c[v] * sScale[mm] + sBias[mm]);
            sYout[col * 128 + mm] = (_Float16)val;
        }
    }
}

// ---------------------------------------------------------------------------
// Kernel 3: fused group -> 3x(conv1x1+BN+ReLU) -> maxpool(ns).
// 256 threads (8 waves), 64 grouped columns per block, everything in LDS.
// Weights arrive pre-converted f16 (stride 128): pure B128 global->LDS copies.
// ---------------------------------------------------------------------------
template<int NS, int C2, int CH_OFF>
__global__ __launch_bounds__(256) void group_mlp_kernel(
    const float* __restrict__ xyz,  const float* __restrict__ feat,
    const float* __restrict__ new_xyz, const int* __restrict__ idx,
    const _Float16* __restrict__ Wh,     // [64*128 | C2*128 | 128*128]
    const float* __restrict__ scl,       // [64 | C2 | 128]
    const float* __restrict__ bia,       // [64 | C2 | 128]
    float* __restrict__ out_feat)
{
    constexpr int COLS = 64;
    constexpr int PTS  = COLS / NS;
    constexpr int C3   = 128;
    constexpr int K0   = 96;             // 67 padded to 3x32 (zeros pre-baked)
    constexpr int OFF2 = 64 * 128;
    constexpr int OFF3 = OFF2 + C2 * 128;

    __shared__ __attribute__((aligned(32))) _Float16 sW[C3 * 128];    // 32 KB
    __shared__ __attribute__((aligned(32))) _Float16 sX[COLS * 128];  // 16 KB
    __shared__ __attribute__((aligned(32))) _Float16 sY[COLS * 128];  // 16 KB
    __shared__ float sScale[C3], sBias[C3];
    __shared__ int   sIdx[COLS];
    __shared__ float sNx[PTS][3];

    const int b    = blockIdx.y;
    const int p0   = blockIdx.x * PTS;
    const int tid  = threadIdx.x;
    const int lane = tid & 31, wv = tid >> 5;

    // --- stage 0: gather indices, centroids, build X0 (rel_xyz ++ features) ---
    if (tid < COLS)
        sIdx[tid] = idx[((size_t)b*NPOINT_ + p0) * NS + tid];
    if (tid < PTS*3) {
        int p = tid / 3, d = tid % 3;
        sNx[p][d] = new_xyz[((size_t)b*NPOINT_ + p0 + p)*3 + d];
    }
    __syncthreads();
    for (int e = tid; e < COLS * K0; e += 256) {
        const int j = e % COLS, k = e / COLS;
        const int i = sIdx[j];
        float v;
        if (k < 3)       v = xyz[((size_t)b*N_ + i)*3 + k] - sNx[j / NS][k];
        else if (k < 67) v = feat[((size_t)b*C_ + (k-3))*N_ + i];
        else             v = 0.0f;
        sX[j * 128 + k] = (_Float16)v;
    }

    // --- layer 1: 67(->96) -> 64 ---
    {
        const v16h* src = (const v16h*)(Wh);
        v16h* dst = (v16h*)sW;
        for (int e = tid; e < 64 * 8; e += 256) dst[e] = src[e];
        if (tid < 64) { sScale[tid] = scl[tid]; sBias[tid] = bia[tid]; }
    }
    __syncthreads();
    mlp_layer(sX, sY, sW, sScale, sBias, 64, K0, lane, wv);
    __syncthreads();

    // --- layer 2: 64 -> C2 ---
    {
        const v16h* src = (const v16h*)(Wh + OFF2);
        v16h* dst = (v16h*)sW;
        for (int e = tid; e < C2 * 8; e += 256) dst[e] = src[e];
        if (tid < C2) { sScale[tid] = scl[64 + tid]; sBias[tid] = bia[64 + tid]; }
    }
    __syncthreads();
    mlp_layer(sY, sX, sW, sScale, sBias, C2, 64, lane, wv);
    __syncthreads();

    // --- layer 3: C2 -> 128 ---
    {
        const v16h* src = (const v16h*)(Wh + OFF3);
        v16h* dst = (v16h*)sW;
        for (int e = tid; e < C3 * 8; e += 256) dst[e] = src[e];
        if (tid < C3) { sScale[tid] = scl[64 + C2 + tid]; sBias[tid] = bia[64 + C2 + tid]; }
    }
    __syncthreads();
    mlp_layer(sX, sY, sW, sScale, sBias, C3, C2, lane, wv);
    __syncthreads();

    // --- maxpool over ns, write (B,256,NPOINT) slice at CH_OFF ---
    for (int e = tid; e < PTS * C3; e += 256) {
        const int p = e / C3, ch = e % C3;
        float mx = -INFINITY;
        #pragma unroll 4
        for (int n = 0; n < NS; ++n)
            mx = fmaxf(mx, (float)sY[(p*NS + n) * 128 + ch]);
        out_feat[((size_t)b*256 + CH_OFF + ch)*NPOINT_ + (p0 + p)] = mx;
    }
}

// ---------------------------------------------------------------------------
extern "C" void kernel_launch(void* const* d_in, const int* in_sizes, int n_in,
                              void* d_out, int out_size, void* d_ws, size_t ws_size,
                              hipStream_t stream)
{
    const float* xyz  = (const float*)d_in[0];
    const float* feat = (const float*)d_in[1];
    // params0: d_in[2..10] = W,g,b x3 ; params1: d_in[11..19]
    const float* P[18];
    for (int i = 0; i < 18; ++i) P[i] = (const float*)d_in[2 + i];

    float* out      = (float*)d_out;
    float* new_xyz  = out;                                  // (B, 2048, 3)
    float* out_feat = out + (size_t)B_ * NPOINT_ * 3;       // (B, 256, 2048)

    // ---- workspace layout ----
    char* w = (char*)d_ws;
    int*      fpsIdx = (int*)w;                 w += (size_t)B_*NPOINT_*4;
    int*      idx0   = (int*)w;                 w += (size_t)B_*NPOINT_*16*4;
    int*      idx1   = (int*)w;                 w += (size_t)B_*NPOINT_*32*4;
    _Float16* wh0    = (_Float16*)w;            w += (size_t)(64+64+128)*128*2;
    _Float16* wh1    = (_Float16*)w;            w += (size_t)(64+128+128)*128*2;
    float*    scl0   = (float*)w;               w += (size_t)(64+64+128)*4;
    float*    bia0   = (float*)w;               w += (size_t)(64+64+128)*4;
    float*    scl1   = (float*)w;               w += (size_t)(64+128+128)*4;
    float*    bia1   = (float*)w;               w += (size_t)(64+128+128)*4;

    // ---- one-time weight prep (f32 -> padded f16, BN folded) ----
    struct { const float *W, *g, *b; int Co, Ci; _Float16* dW; float *dS, *dB; }
    L[6] = {
        { P[0],  P[1],  P[2],  64,  67, wh0,            scl0,          bia0          },
        { P[3],  P[4],  P[5],  64,  64, wh0 + 64*128,   scl0 + 64,     bia0 + 64     },
        { P[6],  P[7],  P[8],  128, 64, wh0 + 128*128,  scl0 + 128,    bia0 + 128    },
        { P[9],  P[10], P[11], 64,  67, wh1,            scl1,          bia1          },
        { P[12], P[13], P[14], 128, 64, wh1 + 64*128,   scl1 + 64,     bia1 + 64     },
        { P[15], P[16], P[17], 128, 128, wh1 + 192*128, scl1 + 192,    bia1 + 192    },
    };
    for (int i = 0; i < 6; ++i)
        prep_weights_kernel<<<(L[i].Co*128 + 255)/256, 256, 0, stream>>>(
            L[i].W, L[i].g, L[i].b, L[i].Co, L[i].Ci, L[i].dW, L[i].dS, L[i].dB);

    // ---- pipeline ----
    fps_kernel<<<B_, 1024, 0, stream>>>(xyz, new_xyz, fpsIdx);

    const int bq_threads = B_ * NPOINT_ * 32;               // one wave per query
    ball_query_kernel<16><<<(bq_threads + 255)/256, 256, 0, stream>>>(
        xyz, new_xyz, idx0, 0.1f);
    ball_query_kernel<32><<<(bq_threads + 255)/256, 256, 0, stream>>>(
        xyz, new_xyz, idx1, 0.2f);

    group_mlp_kernel<16, 64, 0><<<dim3(NPOINT_/4, B_), 256, 0, stream>>>(
        xyz, feat, new_xyz, idx0, wh0, scl0, bia0, out_feat);

    group_mlp_kernel<32, 128, 128><<<dim3(NPOINT_/2, B_), 256, 0, stream>>>(
        xyz, feat, new_xyz, idx1, wh1, scl1, bia1, out_feat);
}
